// exp_acq_46531675685182
// MI455X (gfx1250) — matrix operand order repaired and checked by reading, not compiled, not run on hardware
//
#include <hip/hip_runtime.h>
#include <stdint.h>

// ---------------------------------------------------------------------------
// Fused (3x GEMM + activations + GRU-like scan) for MI455X / gfx1250.
//   out[b,0,:]   = exp
//   out[b,t+1,:] = r_t * h + (1-r_t) * (sigmoid(xWg+bg)*tanh(xWk+bk)),  t=0..98
//   x[b,t,:] = concat(qa[b,t,:], q[b,t,:])            (K = 512)
// bf16 WMMA (f32 accum), weights staged in LDS as bf16, X streamed via
// global_load_async_to_lds_b128 with double buffering.
// f32->bf16 now via hardware v_cvt_pk_bf16_f32 (1 VALU per pair).
// ---------------------------------------------------------------------------

typedef __attribute__((ext_vector_type(16))) __bf16 v16bf;
typedef __attribute__((ext_vector_type(8)))  float  v8f;

union BF16x16 { uint32_t u[8]; v16bf v; };

#define BATCH 4096
#define LSEQ  100
#define DDIM  256
#define SEQ   99
#define KDIM  512

#define W_LDS   (3 * 64 * 1024)     // 196608 B : bf16 weights, 3 mats x 64 rows
#define XBUF    (32 * 2048)         //  65536 B : one X tile (32 rows x 512 f32)
#define LDS_TOT (W_LDS + 2 * XBUF)  // 327680 B = 320 KB (full WGP LDS)

// f32 pair -> packed bf16 (RNE) in one VALU op.
__device__ __forceinline__ uint32_t pk_bf16(float a, float b) {
  uint32_t d;
  asm("v_cvt_pk_bf16_f32 %0, %1, %2" : "=v"(d) : "v"(a), "v"(b));
  return d;
}

// CDNA5 async global->LDS copy, 16B per lane, tracked by ASYNCcnt
__device__ __forceinline__ void async_cp16(uint32_t lds_off, const void* gsrc) {
  asm volatile("global_load_async_to_lds_b128 %0, %1, off"
               :: "v"(lds_off), "v"(gsrc) : "memory");
}
__device__ __forceinline__ void wait_async0() {
  asm volatile("s_wait_asynccnt 0" ::: "memory");
}

__device__ __forceinline__ float fast_sig(float x) {
  return __fdividef(1.0f, 1.0f + __expf(-x));
}

// Stage X tile for time t into LDS (f32, 16B chunks XOR-swizzled by row&15).
// 32 rows x 128 chunks = 4096 chunks; 256 threads x 16 chunks each.
__device__ __forceinline__ void prefetch_x(const float* __restrict__ qa,
                                           const float* __restrict__ q,
                                           char* xbase, int b0, int t, int tid) {
#pragma unroll
  for (int j = 0; j < 16; ++j) {
    const int id = tid + (j << 8);
    const int r  = id >> 7;      // local batch row 0..31
    const int c  = id & 127;     // 16B chunk in 512-f32 row; c<64 -> qa half
    const float* src = ((c < 64) ? qa : q)
                     + ((size_t)(b0 + r) * LSEQ + t) * DDIM + ((c & 63) << 2);
    const uint32_t lds =
        (uint32_t)(uintptr_t)(xbase + r * 2048 + ((c ^ (r & 15)) << 4));
    async_cp16(lds, src);
  }
}

__global__ __launch_bounds__(256) void exp_acq_fused(
    const float* __restrict__ qemb, const float* __restrict__ qaemb,
    const float* __restrict__ expv,
    const float* __restrict__ Wg, const float* __restrict__ bgp,
    const float* __restrict__ Wk, const float* __restrict__ bkp,
    const float* __restrict__ Wr, const float* __restrict__ brp,
    float* __restrict__ out)
{
  extern __shared__ char smem[];
  char* xbuf0 = smem + W_LDS;
  char* xbuf1 = smem + W_LDS + XBUF;

  const int tid  = threadIdx.x;
  const int lane = tid & 31;
  const int w    = tid >> 5;     // wave 0..7
  const int w4   = w & 3;        // column subtile (16 cols)
  const int bgrp = w >> 2;       // batch subtile (16 rows)
  const int hsel = lane >> 4;    // half-wave select (K/row-half per ISA layout)
  const int nbase = blockIdx.x * 64;
  const int b0    = blockIdx.y * 32;

  // Kick off async prefetch of X tile t=0 before touching weights.
  prefetch_x(qaemb, qemb, xbuf0, b0, 0, tid);

  // Weights f32 -> bf16 into LDS, 16B chunks swizzled by (row&15).
  for (int idx = tid; idx < 3 * 64 * 64; idx += 256) {
    const int mat = idx >> 12;
    const int row = (idx >> 6) & 63;
    const int kc  = idx & 63;
    const float* wsel = (mat == 0) ? Wg : ((mat == 1) ? Wk : Wr);
    const float* src  = wsel + ((size_t)(nbase + row) << 9) + (kc << 3);
    const float4 f0 = *(const float4*)(src);
    const float4 f1 = *(const float4*)(src + 4);
    uint4 pk;
    pk.x = pk_bf16(f0.x, f0.y); pk.y = pk_bf16(f0.z, f0.w);
    pk.z = pk_bf16(f1.x, f1.y); pk.w = pk_bf16(f1.z, f1.w);
    *(uint4*)(smem + mat * 65536 + row * 1024 + ((kc ^ (row & 15)) << 4)) = pk;
  }

  // Per-lane constants (column = dcol for both half-waves).
  const int   dcol = nbase + w4 * 16 + (lane & 15);
  const float ev   = expv[dcol];
  const float bgc  = bgp[dcol];
  const float bkc  = bkp[dcol];
  const float brc  = brp[dcol];

  const int xrow = bgrp * 16 + (lane & 15);   // A-fragment source row (batch)
  const int xs   = xrow & 15;
  const int wrow = w4 * 16 + (lane & 15);     // B-fragment source row (out col)
  const int ws   = wrow & 15;
  const char* wp0 = smem + 0 * 65536 + wrow * 1024;
  const char* wp1 = smem + 1 * 65536 + wrow * 1024;
  const char* wp2 = smem + 2 * 65536 + wrow * 1024;

  // C/D layout: element i -> batch row (hsel*8 + i), column dcol.
  float* outp = out + (size_t)(b0 + bgrp * 16 + hsel * 8) * LSEQ * DDIM + dcol;
  float h[8];
#pragma unroll
  for (int i = 0; i < 8; ++i) {
    h[i] = ev;
    outp[(size_t)i * LSEQ * DDIM] = ev;      // out[:, 0, :] = exp
  }

  wait_async0();
  __syncthreads();

  for (int t = 0; t < SEQ; ++t) {
    if (t + 1 < SEQ)  // overlap next X tile with this step's WMMA
      prefetch_x(qaemb, qemb, ((t + 1) & 1) ? xbuf1 : xbuf0, b0, t + 1, tid);

    const char* xrowp = (((t & 1) ? xbuf1 : xbuf0)) + xrow * 2048;

    v8f accG = {0.f,0.f,0.f,0.f,0.f,0.f,0.f,0.f};
    v8f accK = accG;
    v8f accR = accG;

#pragma unroll
    for (int kk = 0; kk < 16; ++kk) {
      // --- A fragment: row xrow, 16 contiguous K (k = 16*hsel + s), matching B ---
      const int ec = kk * 8 + hsel * 4;  // 16B f32 chunk index (4 f32 each)
      const float4 xa0 = *(const float4*)(xrowp + (((ec    ) ^ xs) << 4));
      const float4 xa1 = *(const float4*)(xrowp + (((ec + 1) ^ xs) << 4));
      const float4 xa2 = *(const float4*)(xrowp + (((ec + 2) ^ xs) << 4));
      const float4 xa3 = *(const float4*)(xrowp + (((ec + 3) ^ xs) << 4));
      BF16x16 af;
      af.u[0] = pk_bf16(xa0.x, xa0.y); af.u[1] = pk_bf16(xa0.z, xa0.w);
      af.u[2] = pk_bf16(xa1.x, xa1.y); af.u[3] = pk_bf16(xa1.z, xa1.w);
      af.u[4] = pk_bf16(xa2.x, xa2.y); af.u[5] = pk_bf16(xa2.z, xa2.w);
      af.u[6] = pk_bf16(xa3.x, xa3.y); af.u[7] = pk_bf16(xa3.z, xa3.w);

      // --- B fragments: 16 contiguous K of W row wrow (two 16B chunks) ---
      const int kc = kk * 4 + hsel * 2;  // 16B bf16 chunk index (8 vals each)
      {
        const uint4 c0 = *(const uint4*)(wp0 + (((kc    ) ^ ws) << 4));
        const uint4 c1 = *(const uint4*)(wp0 + (((kc + 1) ^ ws) << 4));
        BF16x16 bf;
        bf.u[0]=c0.x; bf.u[1]=c0.y; bf.u[2]=c0.z; bf.u[3]=c0.w;
        bf.u[4]=c1.x; bf.u[5]=c1.y; bf.u[6]=c1.z; bf.u[7]=c1.w;
        accG = __builtin_amdgcn_wmma_f32_16x16x32_bf16(false, af.v, false, bf.v,
                                                       (short)0, accG, false, false);
      }
      {
        const uint4 c0 = *(const uint4*)(wp1 + (((kc    ) ^ ws) << 4));
        const uint4 c1 = *(const uint4*)(wp1 + (((kc + 1) ^ ws) << 4));
        BF16x16 bf;
        bf.u[0]=c0.x; bf.u[1]=c0.y; bf.u[2]=c0.z; bf.u[3]=c0.w;
        bf.u[4]=c1.x; bf.u[5]=c1.y; bf.u[6]=c1.z; bf.u[7]=c1.w;
        accK = __builtin_amdgcn_wmma_f32_16x16x32_bf16(false, af.v, false, bf.v,
                                                       (short)0, accK, false, false);
      }
      {
        const uint4 c0 = *(const uint4*)(wp2 + (((kc    ) ^ ws) << 4));
        const uint4 c1 = *(const uint4*)(wp2 + (((kc + 1) ^ ws) << 4));
        BF16x16 bf;
        bf.u[0]=c0.x; bf.u[1]=c0.y; bf.u[2]=c0.z; bf.u[3]=c0.w;
        bf.u[4]=c1.x; bf.u[5]=c1.y; bf.u[6]=c1.z; bf.u[7]=c1.w;
        accR = __builtin_amdgcn_wmma_f32_16x16x32_bf16(false, af.v, false, bf.v,
                                                       (short)0, accR, false, false);
      }
    }

    // Epilogue: activations + recurrence, h stays in C-layout registers.
#pragma unroll
    for (int i = 0; i < 8; ++i) {
      const float gt = fast_sig(accG[i] + bgc);
      const float kt = 2.0f * fast_sig(2.0f * (accK[i] + bkc)) - 1.0f;  // tanh
      const float rt = fast_sig(accR[i] + brc);
      h[i] = rt * h[i] + (1.0f - rt) * (gt * kt);
      outp[((size_t)i * LSEQ + (t + 1)) * DDIM] = h[i];
    }

    wait_async0();     // prefetch of t+1 finished
    __syncthreads();   // everyone done reading buf(t&1) / writing buf(t+1&1)
  }
}

extern "C" void kernel_launch(void* const* d_in, const int* in_sizes, int n_in,
                              void* d_out, int out_size, void* d_ws, size_t ws_size,
                              hipStream_t stream) {
  (void)in_sizes; (void)n_in; (void)out_size; (void)d_ws; (void)ws_size;
  const float* q   = (const float*)d_in[0];
  const float* qa  = (const float*)d_in[1];
  const float* ex  = (const float*)d_in[2];
  const float* Wg  = (const float*)d_in[3];
  const float* bg  = (const float*)d_in[4];
  const float* Wk  = (const float*)d_in[5];
  const float* bk  = (const float*)d_in[6];
  const float* Wr  = (const float*)d_in[7];
  const float* br  = (const float*)d_in[8];

  static int attr_set = 0;  // idempotent, deterministic
  if (!attr_set) {
    hipFuncSetAttribute((const void*)exp_acq_fused,
                        hipFuncAttributeMaxDynamicSharedMemorySize, LDS_TOT);
    attr_set = 1;
  }

  // grid.x = column tiles (fast-varying so the 4 WGs sharing an X row
  // co-schedule and L2 captures the 4x reuse of q/qa).
  dim3 grid(DDIM / 64, BATCH / 32);
  exp_acq_fused<<<grid, dim3(256), LDS_TOT, stream>>>(
      q, qa, ex, Wg, bg, Wk, bk, Wr, br, (float*)d_out);
}
